// LinearAttention_35734127903047
// MI455X (gfx1250) — matrix operand order repaired and checked by reading, not compiled, not run on hardware
//
#include <hip/hip_runtime.h>
#include <hip/hip_bf16.h>
#include <math.h>

#define D 512
#define BD (D * D)
#define NROWS (4 * 8192)   // B*S

typedef _Float16 v16h __attribute__((ext_vector_type(16)));
typedef _Float16 h4   __attribute__((ext_vector_type(4)));
typedef float    v8f  __attribute__((ext_vector_type(8)));

union FragU { v16h v; uint4 u[2]; };
union H4U   { uint2 u; _Float16 h[4]; };

// A-matrix 16x32 f16 fragment from a 32-element (64B, 16B-aligned) row.
// Lanes 0-15 hold K{0..15}; lanes 16-31 hold K{16..31}.
__device__ __forceinline__ v16h wload_a(const _Float16* row32, int hi) {
  const uint4* p = (const uint4*)row32;
  FragU f;
  f.u[0] = p[2 * hi];
  f.u[1] = p[2 * hi + 1];
  return f.v;
}

// B-matrix 32x16 f16 fragment: lane n reads 16 contiguous K values (32B).
__device__ __forceinline__ v16h wload_b(const _Float16* k16) {
  const uint4* p = (const uint4*)k16;
  FragU f;
  f.u[0] = p[0];
  f.u[1] = p[1];
  return f.v;
}

#define WMMA(a, b, c) \
  __builtin_amdgcn_wmma_f32_16x16x32_f16(false, (a), false, (b), (short)0, (c), false, false)

// ---------------------------------------------------------------- pass 0
__global__ void cvt_f32_f16(const float* __restrict__ src,
                            _Float16* __restrict__ dst, int n4) {
  int i = blockIdx.x * blockDim.x + threadIdx.x;
  if (i < n4) {
    float4 f = ((const float4*)src)[i];
    H4U h;
    h.h[0] = (_Float16)f.x; h.h[1] = (_Float16)f.y;
    h.h[2] = (_Float16)f.z; h.h[3] = (_Float16)f.w;
    ((uint2*)dst)[i] = h.u;
  }
}

// ---------------------------------------------------------------- pass 1
// q = elu1(x Wq^T + bq), k = elu1(x Wk^T + bk), v = x Wv^T + bv  (f16 out)
// Block: 32 rows x 48 col-tiles (half of the 96 q|k|v col-tiles).
#define XPAD 520   // 512 + 8 f16 pad: 1040B row keeps 16B align, breaks bank collisions
__global__ void __launch_bounds__(256, 1)
qkv_proj(const float* __restrict__ x,
         const _Float16* __restrict__ Wall,   // Wq|Wk|Wv contiguous, 3*BD
         const float* __restrict__ bq, const float* __restrict__ bk,
         const float* __restrict__ bv,
         _Float16* __restrict__ qo, _Float16* __restrict__ ko,
         _Float16* __restrict__ vo) {
  __shared__ __align__(16) _Float16 xs[32 * XPAD];
  const int t = threadIdx.x;
  const int cg = blockIdx.x & 1;              // column group
  const int row0 = (blockIdx.x >> 1) * 32;

  // stage x tile (32 x 512) to LDS as f16
#pragma unroll
  for (int i = 0; i < 16; ++i) {
    int lin = t + i * 256;                    // 0..4095 float4 chunks
    int r = lin >> 7, c4 = (lin & 127) << 2;
    float4 f = *(const float4*)(x + (size_t)(row0 + r) * D + c4);
    h4 h;
    h.x = (_Float16)f.x; h.y = (_Float16)f.y;
    h.z = (_Float16)f.z; h.w = (_Float16)f.w;
    *(h4*)&xs[r * XPAD + c4] = h;
  }
  __syncthreads();

  const int w = t >> 5, lane = t & 31;
  const int n = lane & 15, hi = lane >> 4;
  const int loff = n * D + hi * 16;           // lane-local element offset (32-bit)

  const v8f zero8 = {0.f, 0.f, 0.f, 0.f, 0.f, 0.f, 0.f, 0.f};
  v8f acc0[6], acc1[6];
  const _Float16* wbase[6];                   // uniform (SGPR) bases
  float bsel[6];
  int colb[6], sel[6];
#pragma unroll
  for (int j = 0; j < 6; ++j) {
    acc0[j] = zero8; acc1[j] = zero8;
    int tidx = cg * 48 + w * 6 + j;           // 0..95 col tiles across q|k|v
    int s = tidx >> 5;                        // 0=q 1=k 2=v
    int cb = (tidx & 31) << 4;
    wbase[j] = Wall + (size_t)s * BD + (size_t)cb * D;
    const float* bb = (s == 0) ? bq : (s == 1) ? bk : bv;
    bsel[j] = bb[cb + n];
    colb[j] = cb; sel[j] = s;
  }

#pragma unroll 2
  for (int k0 = 0; k0 < D; k0 += 32) {
    v16h a0 = wload_a(&xs[n * XPAD + k0], hi);
    v16h a1 = wload_a(&xs[(16 + n) * XPAD + k0], hi);
    v16h bf[6];
#pragma unroll
    for (int j = 0; j < 6; ++j)               // issue all B loads first
      bf[j] = wload_b(wbase[j] + loff + k0);
#pragma unroll
    for (int j = 0; j < 6; ++j) {             // then all WMMAs (B reused x2)
      acc0[j] = WMMA(a0, bf[j], acc0[j]);
      acc1[j] = WMMA(a1, bf[j], acc1[j]);
    }
  }

#pragma unroll
  for (int j = 0; j < 6; ++j) {
    _Float16* outp = (sel[j] == 0) ? qo : (sel[j] == 1) ? ko : vo;
    float bias = bsel[j];
#pragma unroll
    for (int i = 0; i < 8; ++i) {
      int r = i + hi * 8;
      float v0 = acc0[j][i] + bias;
      float v1 = acc1[j][i] + bias;
      if (sel[j] < 2) {
        v0 = (v0 > 0.f) ? (v0 + 1.f) : __expf(v0);
        v1 = (v1 > 0.f) ? (v1 + 1.f) : __expf(v1);
      }
      outp[(size_t)(row0 + r) * D + colb[j] + n]      = (_Float16)v0;
      outp[(size_t)(row0 + 16 + r) * D + colb[j] + n] = (_Float16)v1;
    }
  }
}

// ---------------------------------------------------------------- pass 2
// KV[b,d,e] = sum_s k[b,s,d] * v[b,s,e]; store transposed KVt[b,e,d] (f16)
// Block tile: 64(d) x 128(e); wave = 2x2 16x16 tiles -> each LDS frag feeds 2 WMMAs.
#define SPAD 40    // 32 + 8: row = 80B, 16B aligned
__global__ void __launch_bounds__(256, 1)
kv_outer(const _Float16* __restrict__ k, const _Float16* __restrict__ v,
         _Float16* __restrict__ KVt) {
  __shared__ __align__(16) _Float16 kT[64 * SPAD];
  __shared__ __align__(16) _Float16 vT[128 * SPAD];
  const int t = threadIdx.x;
  const int tile = blockIdx.x & 31;           // 8(d) x 4(e) tiles
  const int b = blockIdx.x >> 5;
  const int d0 = (tile & 7) * 64;
  const int e0 = (tile >> 3) * 128;
  const _Float16* kb = k + (size_t)b * 8192 * D;
  const _Float16* vb = v + (size_t)b * 8192 * D;

  const int w = t >> 5, lane = t & 31, n = lane & 15, hi = lane >> 4;
  const int wm = w >> 2, wn = w & 3;          // wave grid 2(M) x 4(N)

  const v8f zero8 = {0.f, 0.f, 0.f, 0.f, 0.f, 0.f, 0.f, 0.f};
  v8f acc[2][2] = {{zero8, zero8}, {zero8, zero8}};

  for (int s0 = 0; s0 < 8192; s0 += 32) {
    // stage transposed k chunk [64 d][32 s]
#pragma unroll
    for (int i = 0; i < 2; ++i) {
      int lin = t + i * 256;                  // 0..511 chunks of 4 f16
      int si = lin >> 4, c4 = (lin & 15) << 2;
      H4U hk;
      hk.u = *(const uint2*)(kb + (size_t)(s0 + si) * D + d0 + c4);
#pragma unroll
      for (int m = 0; m < 4; ++m) kT[(c4 + m) * SPAD + si] = hk.h[m];
    }
    // stage transposed v chunk [128 e][32 s]
#pragma unroll
    for (int i = 0; i < 4; ++i) {
      int lin = t + i * 256;                  // 0..1023 chunks of 4 f16
      int si = lin >> 5, c4 = (lin & 31) << 2;
      H4U hv;
      hv.u = *(const uint2*)(vb + (size_t)(s0 + si) * D + e0 + c4);
#pragma unroll
      for (int m = 0; m < 4; ++m) vT[(c4 + m) * SPAD + si] = hv.h[m];
    }
    if (s0 + 32 < 8192) {                     // gfx1250 global_prefetch_b8
      __builtin_prefetch(kb + (size_t)(s0 + 32 + (t >> 3)) * D + d0, 0, 1);
      __builtin_prefetch(vb + (size_t)(s0 + 32 + (t >> 3)) * D + e0, 0, 1);
    }
    __syncthreads();

    v16h a0 = wload_a(&kT[(wm * 32 + n) * SPAD], hi);
    v16h a1 = wload_a(&kT[(wm * 32 + 16 + n) * SPAD], hi);
    v16h b0 = wload_b(&vT[(wn * 32 + n) * SPAD + hi * 16]);
    v16h b1 = wload_b(&vT[(wn * 32 + 16 + n) * SPAD + hi * 16]);
    acc[0][0] = WMMA(a0, b0, acc[0][0]);
    acc[1][0] = WMMA(a1, b0, acc[1][0]);
    acc[0][1] = WMMA(a0, b1, acc[0][1]);
    acc[1][1] = WMMA(a1, b1, acc[1][1]);
    __syncthreads();
  }

  _Float16* outp = KVt + (size_t)b * BD;
#pragma unroll
  for (int jn = 0; jn < 2; ++jn) {
    int e = e0 + wn * 32 + jn * 16 + n;
#pragma unroll
    for (int jm = 0; jm < 2; ++jm)
#pragma unroll
      for (int i = 0; i < 8; ++i) {
        int d = d0 + wm * 32 + jm * 16 + i + hi * 8;
        outp[(size_t)e * D + d] = (_Float16)acc[jm][jn][i];
      }
  }
}

// ---------------------------------------------------------------- pass 3
// attn = q @ KV, then out = layernorm(x + attn)
__global__ void __launch_bounds__(256, 1)
attn_ln(const _Float16* __restrict__ q, const _Float16* __restrict__ KVt,
        const float* __restrict__ x, const float* __restrict__ gamma,
        const float* __restrict__ beta, float* __restrict__ out) {
  __shared__ __align__(16) _Float16 qs[16 * XPAD];
  __shared__ __align__(16) float ys[16 * D];
  const int t = threadIdx.x;
  const int row0 = blockIdx.x * 16;           // global row in [0, 32768)
  const int b = row0 >> 13;
  const _Float16* KV = KVt + (size_t)b * BD;

  // stage q tile (already f16): uint4 = 8 halves
#pragma unroll
  for (int i = 0; i < 4; ++i) {
    int lin = t + i * 256;                    // 0..1023 chunks
    int r = lin >> 6, c8 = (lin & 63) << 3;
    *(uint4*)&qs[r * XPAD + c8] =
        *(const uint4*)(q + (size_t)(row0 + r) * D + c8);
  }
  __syncthreads();

  const int w = t >> 5, lane = t & 31, n = lane & 15, hi = lane >> 4;
  const int loff = n * D + hi * 16;
  const int cb0 = w * 64;                     // wave owns 4 col-tiles (64 cols)
  const v8f zero8 = {0.f, 0.f, 0.f, 0.f, 0.f, 0.f, 0.f, 0.f};
  v8f acc[4] = {zero8, zero8, zero8, zero8};
  const _Float16* kvb[4];                     // uniform (SGPR) bases
#pragma unroll
  for (int j = 0; j < 4; ++j) kvb[j] = KV + (size_t)(cb0 + j * 16) * D;

#pragma unroll 2
  for (int k0 = 0; k0 < D; k0 += 32) {
    v16h a = wload_a(&qs[n * XPAD + k0], hi);
    v16h bf[4];
#pragma unroll
    for (int j = 0; j < 4; ++j) bf[j] = wload_b(kvb[j] + loff + k0);
#pragma unroll
    for (int j = 0; j < 4; ++j) acc[j] = WMMA(a, bf[j], acc[j]);
  }

  // spill attn tile to LDS for cross-wave row stats
#pragma unroll
  for (int j = 0; j < 4; ++j)
#pragma unroll
    for (int i = 0; i < 8; ++i)
      ys[(i + hi * 8) * D + cb0 + j * 16 + n] = acc[j][i];
  __syncthreads();

  // layernorm: wave w -> rows 2w, 2w+1; lane owns 16 contiguous cols
#pragma unroll
  for (int rr = 0; rr < 2; ++rr) {
    int r = w * 2 + rr;
    float yv[16];
    const float* xr = x + (size_t)(row0 + r) * D + lane * 16;
    float s = 0.f, sq = 0.f;
#pragma unroll
    for (int i = 0; i < 16; i += 4) {
      float4 xv = *(const float4*)(xr + i);
      float4 av = *(const float4*)&ys[r * D + lane * 16 + i];
      yv[i + 0] = xv.x + av.x; yv[i + 1] = xv.y + av.y;
      yv[i + 2] = xv.z + av.z; yv[i + 3] = xv.w + av.w;
#pragma unroll
      for (int m = 0; m < 4; ++m) {
        s += yv[i + m];
        sq += yv[i + m] * yv[i + m];
      }
    }
#pragma unroll
    for (int off = 16; off >= 1; off >>= 1) {  // wave32 reduction
      s  += __shfl_xor(s,  off, 32);
      sq += __shfl_xor(sq, off, 32);
    }
    float mu = s * (1.f / 512.f);
    float var = sq * (1.f / 512.f) - mu * mu;
    float rs = rsqrtf(var + 1e-5f);
    float* orow = out + (size_t)(row0 + r) * D + lane * 16;
    const float* g = gamma + lane * 16;
    const float* be = beta + lane * 16;
#pragma unroll
    for (int i = 0; i < 16; ++i)
      orow[i] = (yv[i] - mu) * rs * g[i] + be[i];
  }
}

// ---------------------------------------------------------------- launch
extern "C" void kernel_launch(void* const* d_in, const int* in_sizes, int n_in,
                              void* d_out, int out_size, void* d_ws,
                              size_t ws_size, hipStream_t stream) {
  const float* x     = (const float*)d_in[0];
  const float* Wq    = (const float*)d_in[1];
  const float* bq    = (const float*)d_in[2];
  const float* Wk    = (const float*)d_in[3];
  const float* bk    = (const float*)d_in[4];
  const float* Wv    = (const float*)d_in[5];
  const float* bv    = (const float*)d_in[6];
  const float* gamma = (const float*)d_in[7];
  const float* beta  = (const float*)d_in[8];
  float* out = (float*)d_out;

  char* ws = (char*)d_ws;
  size_t off = 0;
  auto carve = [&](size_t bytes) -> void* {
    void* p = ws + off;
    off = (off + bytes + 255) & ~(size_t)255;
    return p;
  };
  // Wq|Wk|Wv f16 as ONE contiguous block so tile bases fold into saddr+imm
  _Float16* Wall = (_Float16*)carve((size_t)3 * BD * 2);
  _Float16* qh   = (_Float16*)carve((size_t)NROWS * D * 2);
  _Float16* kh   = (_Float16*)carve((size_t)NROWS * D * 2);
  _Float16* vh   = (_Float16*)carve((size_t)NROWS * D * 2);
  _Float16* KVt  = (_Float16*)carve((size_t)4 * BD * 2);

  cvt_f32_f16<<<BD / 4 / 256, 256, 0, stream>>>(Wq, Wall,          BD / 4);
  cvt_f32_f16<<<BD / 4 / 256, 256, 0, stream>>>(Wk, Wall + BD,     BD / 4);
  cvt_f32_f16<<<BD / 4 / 256, 256, 0, stream>>>(Wv, Wall + 2 * BD, BD / 4);

  qkv_proj<<<(NROWS / 32) * 2, 256, 0, stream>>>(x, Wall, bq, bk, bv,
                                                 qh, kh, vh);
  kv_outer<<<4 * 32, 256, 0, stream>>>(kh, vh, KVt);
  attn_ln<<<NROWS / 16, 256, 0, stream>>>(qh, KVt, x, gamma, beta, out);
}
